// ResGCNModule_74354473828490
// MI455X (gfx1250) — matrix operand order repaired and checked by reading, not compiled, not run on hardware
//
#include <hip/hip_runtime.h>
#include <hip/hip_bf16.h>
#include <math.h>

// ---------------------------------------------------------------------------
// ResGCN on MI455X (gfx1250, wave32). One point per wave: K=16 neighbors map
// to the M=16 dim of v_wmma_f32_16x16x32_f16. Weights staged (pre-swizzled)
// in LDS; per-wave LDS scratch for A-matrix staging between MLP sublayers.
// Attention/softmax done in registers with __shfl_xor over the WMMA D layout.
// ---------------------------------------------------------------------------

typedef __attribute__((ext_vector_type(16))) _Float16 v16h;
typedef __attribute__((ext_vector_type(8)))  _Float16 v8h;
typedef __attribute__((ext_vector_type(8)))  float    v8f;

#define WPB   4        // waves per block (128 threads)
#define NPTS  20000
#define NBAT  2
#define TOTAL (NBAT*NPTS)

// A-fragment (16x32 f16): lane = row M, halves e=0..15 -> K = 16*laneHi + e.
// Two 16B LDS loads.
__device__ __forceinline__ v16h lds_afrag(const _Float16* rowchunk, int laneHi) {
  const _Float16* p = rowchunk + laneHi * 16;
  v8h lo = *(const v8h*)(p);
  v8h hi = *(const v8h*)(p + 8);
  v16h a;
#pragma unroll
  for (int e = 0; e < 8; e++) { a[e] = lo[e]; a[e + 8] = hi[e]; }
  return a;
}

// B-fragment (32x16 f16): lane = col N, halves = K = 16*laneHi + e.
// Weights stored column-chunk-major in LDS: w[(cc*Cout+col)*32 + k].
__device__ __forceinline__ v16h lds_bfrag(const _Float16* colptr, int laneHi) {
  const _Float16* p = colptr + laneHi * 16;
  v8h lo = *(const v8h*)(p);
  v8h hi = *(const v8h*)(p + 8);
  v16h b;
#pragma unroll
  for (int e = 0; e < 8; e++) { b[e] = lo[e]; b[e + 8] = hi[e]; }
  return b;
}

// B-fragment streamed straight from global (head kernel: weights are L2-hot).
__device__ __forceinline__ v16h gbl_bfrag(const float* __restrict__ W, int Cout,
                                          int cc, int col, int laneHi) {
  const int k0 = cc * 32 + laneHi * 16;
  v16h b;
#pragma unroll
  for (int e = 0; e < 16; e++) b[e] = (_Float16)W[(k0 + e) * Cout + col];
  return b;
}

// DS ops are in-order within a wave; this makes cross-lane LDS RAW safe
// without a workgroup barrier.
__device__ __forceinline__ void wave_lds_fence() {
  asm volatile("s_wait_dscnt 0" ::: "memory");
}

// Attention over the WMMA D layout. h3[q][r]: channel c=q*16+lm (lane),
// neighbor row k = 8*laneHi + r. Computes leaky-relu scores, 16-way softmax,
// weighted channel sum, ELU. Both half-waves end with identical outc[].
__device__ __forceinline__ void attn_head(const v8f h3[8], const float* law,
                                          float abv, int lm, int laneHi,
                                          float outc[8]) {
  float aw1v[8], aw2v[8];
#pragma unroll
  for (int q = 0; q < 8; q++) { int c = q * 16 + lm; aw1v[q] = law[c]; aw2v[q] = law[128 + c]; }
  float t1[8];
#pragma unroll
  for (int r = 0; r < 8; r++) {
    float s = 0.f;
#pragma unroll
    for (int q = 0; q < 8; q++) s += h3[q][r] * aw1v[q];
    t1[r] = s;
  }
#pragma unroll
  for (int m = 1; m < 16; m <<= 1) {
#pragma unroll
    for (int r = 0; r < 8; r++) t1[r] += __shfl_xor(t1[r], m, 32);
  }
  // center term: neighbor row 0 lives in the low half-wave, VGPR 0
  float pc = 0.f;
  if (!laneHi) {
#pragma unroll
    for (int q = 0; q < 8; q++) pc += h3[q][0] * aw2v[q];
  }
#pragma unroll
  for (int m = 1; m < 32; m <<= 1) pc += __shfl_xor(pc, m, 32);

  float sv[8]; float mx = -3.0e38f;
#pragma unroll
  for (int r = 0; r < 8; r++) {
    float v = t1[r] + pc + abv;
    v = v > 0.f ? v : 0.2f * v;         // leaky_relu(0.2)
    sv[r] = v; mx = fmaxf(mx, v);
  }
  mx = fmaxf(mx, __shfl_xor(mx, 16, 32));
  float den = 0.f;
#pragma unroll
  for (int r = 0; r < 8; r++) { sv[r] = __expf(sv[r] - mx); den += sv[r]; }
  den += __shfl_xor(den, 16, 32);
  float inv = 1.f / den;
#pragma unroll
  for (int q = 0; q < 8; q++) {
    float o = 0.f;
#pragma unroll
    for (int r = 0; r < 8; r++) o += sv[r] * h3[q][r];
    o += __shfl_xor(o, 16, 32);         // other half-wave holds rows 8..15
    o *= inv;
    outc[q] = o > 0.f ? o : (__expf(o) - 1.f);   // elu
  }
}

// ---------------------------------------------------------------------------
// Layer 0: feat[16 x 25pad32] -> 32 -> 64 -> 128 (relu each), attention, elu.
// ---------------------------------------------------------------------------
__global__ __launch_bounds__(128) void layer0_kernel(
    const float* __restrict__ x, const float* __restrict__ xyz,
    const int* __restrict__ knn,
    const float* __restrict__ w1, const float* __restrict__ w2,
    const float* __restrict__ w3,
    const float* __restrict__ b1, const float* __restrict__ b2,
    const float* __restrict__ b3,
    const float* __restrict__ aw, const float* __restrict__ ab,
    float* __restrict__ hout) {
  __shared__ __align__(16) _Float16 sw1[32 * 32];       // 25x32 padded
  __shared__ __align__(16) _Float16 sw2[64 * 32];       // 32x64
  __shared__ __align__(16) _Float16 sw3[2 * 128 * 32];  // 64x128 (2 K-chunks)
  __shared__ __align__(16) _Float16 scrA[WPB][16 * 64];
  __shared__ __align__(16) _Float16 scrB[WPB][16 * 64];
  __shared__ float sb1[32], sb2[64], sb3[128], saw[256];
  __shared__ float sab;

  const int tid = threadIdx.x;
  for (int i = tid; i < 32 * 32; i += 128) { int k = i & 31, col = i >> 5; sw1[i] = (_Float16)((k < 25) ? w1[k * 32 + col] : 0.f); }
  for (int i = tid; i < 64 * 32; i += 128) { int k = i & 31, col = i >> 5; sw2[i] = (_Float16)w2[k * 64 + col]; }
  for (int i = tid; i < 2 * 128 * 32; i += 128) { int k = i & 31, col = (i >> 5) & 127, cc = i >> 12; sw3[i] = (_Float16)w3[(cc * 32 + k) * 128 + col]; }
  for (int i = tid; i < 32; i += 128) sb1[i] = b1[i];
  for (int i = tid; i < 64; i += 128) sb2[i] = b2[i];
  for (int i = tid; i < 128; i += 128) sb3[i] = b3[i];
  for (int i = tid; i < 256; i += 128) saw[i] = aw[i];
  if (tid == 0) sab = ab[0];
  __syncthreads();

  const int lane = tid & 31, wave = tid >> 5;
  const int lm = lane & 15, laneHi = lane >> 4;
  _Float16* sA = scrA[wave];
  _Float16* sB = scrB[wave];

  for (int p = blockIdx.x * WPB + wave; p < TOTAL; p += gridDim.x * WPB) {
    const int b = p / NPTS;
    const float cx = xyz[p * 3 + 0], cy = xyz[p * 3 + 1], cz = xyz[p * 3 + 2];
    const int nbr = knn[p * 16 + lm];
    if (!laneHi) {  // build feat row for neighbor lm
      _Float16* row = sA + lm * 32;
      const float* xf = x + (size_t)(b * NPTS + nbr) * 16;
#pragma unroll
      for (int c = 0; c < 16; c++) row[c] = (_Float16)xf[c];
      const float* pn = xyz + (size_t)(b * NPTS + nbr) * 3;
      float px = pn[0], py = pn[1], pz = pn[2];
      float sx = px - cx, sy = py - cy, sz = pz - cz;
      row[16] = (_Float16)px; row[17] = (_Float16)py; row[18] = (_Float16)pz;
      row[19] = (_Float16)sx; row[20] = (_Float16)sy; row[21] = (_Float16)sz;
      row[22] = (_Float16)fabsf(sx); row[23] = (_Float16)fabsf(sy); row[24] = (_Float16)fabsf(sz);
#pragma unroll
      for (int c = 25; c < 32; c++) row[c] = (_Float16)0.f;
    }
    wave_lds_fence();

    // 0a: 16x32 @ 25x32 -> 16x32 (relu) into sB
    {
      v16h a0 = lds_afrag(sA + lm * 32, laneHi);
#pragma unroll
      for (int qc = 0; qc < 2; qc++) {
        const int c = qc * 16 + lm;
        v8f acc; float bias = sb1[c];
#pragma unroll
        for (int r = 0; r < 8; r++) acc[r] = bias;
        v16h bb = lds_bfrag(sw1 + c * 32, laneHi);
        acc = __builtin_amdgcn_wmma_f32_16x16x32_f16(false, a0, false, bb, (short)0, acc, false, false);
#pragma unroll
        for (int r = 0; r < 8; r++) { float v = acc[r]; v = v > 0.f ? v : 0.f; sB[(laneHi * 8 + r) * 32 + c] = (_Float16)v; }
      }
    }
    wave_lds_fence();
    // 0b: 16x32 @ 32x64 -> 16x64 (relu) into sA
    {
      v16h a1 = lds_afrag(sB + lm * 32, laneHi);
#pragma unroll
      for (int qc = 0; qc < 4; qc++) {
        const int c = qc * 16 + lm;
        v8f acc; float bias = sb2[c];
#pragma unroll
        for (int r = 0; r < 8; r++) acc[r] = bias;
        v16h bb = lds_bfrag(sw2 + c * 32, laneHi);
        acc = __builtin_amdgcn_wmma_f32_16x16x32_f16(false, a1, false, bb, (short)0, acc, false, false);
#pragma unroll
        for (int r = 0; r < 8; r++) { float v = acc[r]; v = v > 0.f ? v : 0.f; sA[(laneHi * 8 + r) * 64 + c] = (_Float16)v; }
      }
    }
    wave_lds_fence();
    // 0c: 16x64 @ 64x128 -> 16x128 (relu) in registers
    v8f h3[8];
    {
      v16h a20 = lds_afrag(sA + lm * 64, laneHi);
      v16h a21 = lds_afrag(sA + lm * 64 + 32, laneHi);
#pragma unroll
      for (int qc = 0; qc < 8; qc++) {
        const int c = qc * 16 + lm;
        v8f acc; float bias = sb3[c];
#pragma unroll
        for (int r = 0; r < 8; r++) acc[r] = bias;
        v16h bb0 = lds_bfrag(sw3 + (0 * 128 + c) * 32, laneHi);
        acc = __builtin_amdgcn_wmma_f32_16x16x32_f16(false, a20, false, bb0, (short)0, acc, false, false);
        v16h bb1 = lds_bfrag(sw3 + (1 * 128 + c) * 32, laneHi);
        acc = __builtin_amdgcn_wmma_f32_16x16x32_f16(false, a21, false, bb1, (short)0, acc, false, false);
#pragma unroll
        for (int r = 0; r < 8; r++) { float v = acc[r]; acc[r] = v > 0.f ? v : 0.f; }
        h3[qc] = acc;
      }
    }
    float outc[8];
    attn_head(h3, saw, sab, lm, laneHi, outc);
    if (!laneHi) {
#pragma unroll
      for (int q = 0; q < 8; q++) hout[(size_t)p * 128 + q * 16 + lm] = outc[q];
    }
  }
}

// ---------------------------------------------------------------------------
// Layers 1/2: feat[16 x 137pad160] @ [137x128] (relu), attention, elu, +res.
// ---------------------------------------------------------------------------
__global__ __launch_bounds__(128) void layer12_kernel(
    const float* __restrict__ hprev, const float* __restrict__ xyz,
    const int* __restrict__ knn,
    const float* __restrict__ w, const float* __restrict__ b,
    const float* __restrict__ aw, const float* __restrict__ ab,
    float* __restrict__ hout) {
  __shared__ __align__(16) _Float16 sw[5 * 128 * 32];   // 137x128 padded, 40KB
  __shared__ __align__(16) _Float16 scr[WPB][16 * 160]; // 20KB
  __shared__ float sb[128], saw[256];
  __shared__ float sab;
  const int tid = threadIdx.x;
  for (int i = tid; i < 5 * 128 * 32; i += 128) {
    int k = i & 31, col = (i >> 5) & 127, cc = i >> 12;
    int kg = cc * 32 + k;
    sw[i] = (_Float16)((kg < 137) ? w[kg * 128 + col] : 0.f);
  }
  for (int i = tid; i < 128; i += 128) sb[i] = b[i];
  for (int i = tid; i < 256; i += 128) saw[i] = aw[i];
  if (tid == 0) sab = ab[0];
  __syncthreads();

  const int lane = tid & 31, wave = tid >> 5, lm = lane & 15, laneHi = lane >> 4;
  _Float16* sA = scr[wave];

  for (int p = blockIdx.x * WPB + wave; p < TOTAL; p += gridDim.x * WPB) {
    const int b_ = p / NPTS;
    const float cx = xyz[p * 3 + 0], cy = xyz[p * 3 + 1], cz = xyz[p * 3 + 2];
    const int nbr = knn[p * 16 + lm];
    if (!laneHi) {
      _Float16* row = sA + lm * 160;
      const float4* h4 = (const float4*)(hprev + (size_t)(b_ * NPTS + nbr) * 128);
#pragma unroll
      for (int j = 0; j < 32; j++) {
        float4 v = h4[j];
        row[4 * j + 0] = (_Float16)v.x; row[4 * j + 1] = (_Float16)v.y;
        row[4 * j + 2] = (_Float16)v.z; row[4 * j + 3] = (_Float16)v.w;
      }
      const float* pn = xyz + (size_t)(b_ * NPTS + nbr) * 3;
      float px = pn[0], py = pn[1], pz = pn[2];
      float sx = px - cx, sy = py - cy, sz = pz - cz;
      row[128] = (_Float16)px; row[129] = (_Float16)py; row[130] = (_Float16)pz;
      row[131] = (_Float16)sx; row[132] = (_Float16)sy; row[133] = (_Float16)sz;
      row[134] = (_Float16)fabsf(sx); row[135] = (_Float16)fabsf(sy); row[136] = (_Float16)fabsf(sz);
#pragma unroll
      for (int c = 137; c < 160; c++) row[c] = (_Float16)0.f;
    }
    wave_lds_fence();

    v16h afr[5];
#pragma unroll
    for (int cc = 0; cc < 5; cc++) afr[cc] = lds_afrag(sA + lm * 160 + cc * 32, laneHi);
    v8f h3[8];
#pragma unroll
    for (int qc = 0; qc < 8; qc++) {
      const int c = qc * 16 + lm;
      v8f acc; float bias = sb[c];
#pragma unroll
      for (int r = 0; r < 8; r++) acc[r] = bias;
#pragma unroll
      for (int cc = 0; cc < 5; cc++) {
        v16h bb = lds_bfrag(sw + (cc * 128 + c) * 32, laneHi);
        acc = __builtin_amdgcn_wmma_f32_16x16x32_f16(false, afr[cc], false, bb, (short)0, acc, false, false);
      }
#pragma unroll
      for (int r = 0; r < 8; r++) { float v = acc[r]; acc[r] = v > 0.f ? v : 0.f; }
      h3[qc] = acc;
    }
    float outc[8];
    attn_head(h3, saw, sab, lm, laneHi, outc);
    if (!laneHi) {
#pragma unroll
      for (int q = 0; q < 8; q++) {
        int c = q * 16 + lm;
        hout[(size_t)p * 128 + c] = outc[q] + hprev[(size_t)p * 128 + c];
      }
    }
  }
}

// ---------------------------------------------------------------------------
// Head: per 16-point tile: [16x384] -> 128 (relu) -> 128 (relu) -> 64.
// Weights streamed from global (L2-resident, ~288 KB f32 total).
// ---------------------------------------------------------------------------
__global__ __launch_bounds__(128) void head_kernel(
    const float* __restrict__ h0, const float* __restrict__ h1,
    const float* __restrict__ h2,
    const float* __restrict__ w0, const float* __restrict__ b0,
    const float* __restrict__ w1, const float* __restrict__ b1,
    const float* __restrict__ w2, const float* __restrict__ b2,
    float* __restrict__ out) {
  __shared__ __align__(16) _Float16 scr[WPB][16 * 384];  // 48KB
  const int tid = threadIdx.x, lane = tid & 31, wave = tid >> 5;
  const int lm = lane & 15, laneHi = lane >> 4;
  _Float16* sA = scr[wave];
  const int ntiles = TOTAL / 16;

  for (int t = blockIdx.x * WPB + wave; t < ntiles; t += gridDim.x * WPB) {
    if (!laneHi) {
      const int g = t * 16 + lm;
      _Float16* row = sA + lm * 384;
      const float* srcs[3] = { h0 + (size_t)g * 128, h1 + (size_t)g * 128, h2 + (size_t)g * 128 };
#pragma unroll
      for (int part = 0; part < 3; part++) {
        const float4* s4 = (const float4*)srcs[part];
#pragma unroll
        for (int j = 0; j < 32; j++) {
          float4 v = s4[j];
          _Float16* d = row + part * 128 + 4 * j;
          d[0] = (_Float16)v.x; d[1] = (_Float16)v.y; d[2] = (_Float16)v.z; d[3] = (_Float16)v.w;
        }
      }
    }
    wave_lds_fence();

    // gemm1: 384 -> 128, relu, result re-staged f16 into sA (stride 128)
    {
      v8f accs[8];
#pragma unroll
      for (int qc = 0; qc < 8; qc++) {
        const int c = qc * 16 + lm;
        v8f acc; float bias = b0[c];
#pragma unroll
        for (int r = 0; r < 8; r++) acc[r] = bias;
        for (int cc = 0; cc < 12; cc++) {
          v16h a = lds_afrag(sA + lm * 384 + cc * 32, laneHi);
          v16h bb = gbl_bfrag(w0, 128, cc, c, laneHi);
          acc = __builtin_amdgcn_wmma_f32_16x16x32_f16(false, a, false, bb, (short)0, acc, false, false);
        }
#pragma unroll
        for (int r = 0; r < 8; r++) { float v = acc[r]; acc[r] = v > 0.f ? v : 0.f; }
        accs[qc] = acc;
      }
      wave_lds_fence();
#pragma unroll
      for (int qc = 0; qc < 8; qc++) {
        const int c = qc * 16 + lm;
#pragma unroll
        for (int r = 0; r < 8; r++) sA[(laneHi * 8 + r) * 128 + c] = (_Float16)accs[qc][r];
      }
    }
    wave_lds_fence();
    // gemm2: 128 -> 128, relu, re-stage
    {
      v16h a4[4];
#pragma unroll
      for (int cc = 0; cc < 4; cc++) a4[cc] = lds_afrag(sA + lm * 128 + cc * 32, laneHi);
      v8f accs[8];
#pragma unroll
      for (int qc = 0; qc < 8; qc++) {
        const int c = qc * 16 + lm;
        v8f acc; float bias = b1[c];
#pragma unroll
        for (int r = 0; r < 8; r++) acc[r] = bias;
#pragma unroll
        for (int cc = 0; cc < 4; cc++) {
          v16h bb = gbl_bfrag(w1, 128, cc, c, laneHi);
          acc = __builtin_amdgcn_wmma_f32_16x16x32_f16(false, a4[cc], false, bb, (short)0, acc, false, false);
        }
#pragma unroll
        for (int r = 0; r < 8; r++) { float v = acc[r]; acc[r] = v > 0.f ? v : 0.f; }
        accs[qc] = acc;
      }
      wave_lds_fence();
#pragma unroll
      for (int qc = 0; qc < 8; qc++) {
        const int c = qc * 16 + lm;
#pragma unroll
        for (int r = 0; r < 8; r++) sA[(laneHi * 8 + r) * 128 + c] = (_Float16)accs[qc][r];
      }
    }
    wave_lds_fence();
    // gemm3: 128 -> 64, store f32 output
    {
      v16h a4[4];
#pragma unroll
      for (int cc = 0; cc < 4; cc++) a4[cc] = lds_afrag(sA + lm * 128 + cc * 32, laneHi);
#pragma unroll
      for (int qc = 0; qc < 4; qc++) {
        const int c = qc * 16 + lm;
        v8f acc; float bias = b2[c];
#pragma unroll
        for (int r = 0; r < 8; r++) acc[r] = bias;
#pragma unroll
        for (int cc = 0; cc < 4; cc++) {
          v16h bb = gbl_bfrag(w2, 64, cc, c, laneHi);
          acc = __builtin_amdgcn_wmma_f32_16x16x32_f16(false, a4[cc], false, bb, (short)0, acc, false, false);
        }
#pragma unroll
        for (int r = 0; r < 8; r++) {
          const int g = t * 16 + laneHi * 8 + r;
          out[(size_t)g * 64 + c] = acc[r];
        }
      }
    }
  }
}

// ---------------------------------------------------------------------------
// Launch. Param leaves assumed flattened in jax pytree order (sorted dict
// keys): per layer {attn_b, attn_w, mlp_b[..], mlp_w[..]}, then out_b, out_w.
// ---------------------------------------------------------------------------
extern "C" void kernel_launch(void* const* d_in, const int* in_sizes, int n_in,
                              void* d_out, int out_size, void* d_ws, size_t ws_size,
                              hipStream_t stream) {
  (void)in_sizes; (void)out_size; (void)ws_size;
  if (n_in < 25) return;

  const float* x    = (const float*)d_in[0];
  const float* xyz  = (const float*)d_in[1];
  const int*   knn  = (const int*)d_in[2];
  // layer 0
  const float* l0_ab = (const float*)d_in[3];
  const float* l0_aw = (const float*)d_in[4];
  const float* l0_b1 = (const float*)d_in[5];
  const float* l0_b2 = (const float*)d_in[6];
  const float* l0_b3 = (const float*)d_in[7];
  const float* l0_w1 = (const float*)d_in[8];
  const float* l0_w2 = (const float*)d_in[9];
  const float* l0_w3 = (const float*)d_in[10];
  // layer 1
  const float* l1_ab = (const float*)d_in[11];
  const float* l1_aw = (const float*)d_in[12];
  const float* l1_b  = (const float*)d_in[13];
  const float* l1_w  = (const float*)d_in[14];
  // layer 2
  const float* l2_ab = (const float*)d_in[15];
  const float* l2_aw = (const float*)d_in[16];
  const float* l2_b  = (const float*)d_in[17];
  const float* l2_w  = (const float*)d_in[18];
  // head
  const float* ob0 = (const float*)d_in[19];
  const float* ob1 = (const float*)d_in[20];
  const float* ob2 = (const float*)d_in[21];
  const float* ow0 = (const float*)d_in[22];
  const float* ow1 = (const float*)d_in[23];
  const float* ow2 = (const float*)d_in[24];

  float* h0 = (float*)d_ws;
  float* h1 = h0 + (size_t)TOTAL * 128;
  float* h2 = h1 + (size_t)TOTAL * 128;

  const int knn_stride = TOTAL * 16;  // per-layer KNN slice

  layer0_kernel<<<640, 128, 0, stream>>>(x, xyz, knn,
                                         l0_w1, l0_w2, l0_w3,
                                         l0_b1, l0_b2, l0_b3,
                                         l0_aw, l0_ab, h0);
  layer12_kernel<<<640, 128, 0, stream>>>(h0, xyz, knn + knn_stride,
                                          l1_w, l1_b, l1_aw, l1_ab, h1);
  layer12_kernel<<<640, 128, 0, stream>>>(h1, xyz, knn + 2 * knn_stride,
                                          l2_w, l2_b, l2_aw, l2_ab, h2);
  head_kernel<<<625, 128, 0, stream>>>(h0, h1, h2,
                                       ow0, ob0, ow1, ob1, ow2, ob2,
                                       (float*)d_out);
}
